// DecoderRnn_34780645163081
// MI455X (gfx1250) — matrix operand order repaired and checked by reading, not compiled, not run on hardware
//
#include <hip/hip_runtime.h>
#include <hip/hip_bf16.h>

#define VOCAB  32000
#define HID    512
#define BATCH  32
#define TLEN   64
#define G4H    2048   // 4*HID

typedef __bf16 bf16_t;
typedef __attribute__((ext_vector_type(8)))  __bf16 v8bf;
typedef __attribute__((ext_vector_type(16))) __bf16 v16bf;
typedef __attribute__((ext_vector_type(8)))  float  v8f;

// ---------------------------------------------------------------------------
// helpers
// ---------------------------------------------------------------------------
__device__ __forceinline__ v8f zero8() {
    v8f z;
#pragma unroll
    for (int i = 0; i < 8; ++i) z[i] = 0.0f;
    return z;
}

// A-fragment (16x32 bf16, M x K), row-major source with leading dim ld.
// Lane L holds row (passed in); halves 0..7 -> K = k0 + (L<16?0:8) + 0..7,
// halves 8..15 -> K = k0 + 16 + (L<16?0:8) + 0..7  (CDNA5 ISA 7.12.2).
__device__ __forceinline__ v16bf load_frag_a(const bf16_t* __restrict__ base,
                                             int row, int ld, int k0, int lane) {
    const int kh = (lane < 16) ? 0 : 8;
    const bf16_t* p = base + (long)row * ld + k0 + kh;
    v8bf lo = *reinterpret_cast<const v8bf*>(p);
    v8bf hi = *reinterpret_cast<const v8bf*>(p + 16);
    return __builtin_shufflevector(lo, hi, 0, 1, 2, 3, 4, 5, 6, 7,
                                           8, 9, 10, 11, 12, 13, 14, 15);
}

// B-fragment (32x16 bf16, K x N); weight stored row-major [N_total x K].
// Lane L holds column (weight row passed in); halves 0..7 ->
// K = k0 + (L<16?0:8) + 0..7, halves 8..15 -> K = k0 + 16 + (L<16?0:8) + 0..7.
__device__ __forceinline__ v16bf load_frag_b(const bf16_t* __restrict__ base,
                                             int row, int ld, int k0, int lane) {
    const int kh = (lane < 16) ? 0 : 8;
    const bf16_t* p = base + (long)row * ld + k0 + kh;
    v8bf lo = *reinterpret_cast<const v8bf*>(p);
    v8bf hi = *reinterpret_cast<const v8bf*>(p + 16);
    return __builtin_shufflevector(lo, hi, 0, 1, 2, 3, 4, 5, 6, 7,
                                           8, 9, 10, 11, 12, 13, 14, 15);
}

__device__ __forceinline__ v8f wmma_bf16(v16bf a, v16bf b, v8f c) {
    return __builtin_amdgcn_wmma_f32_16x16x32_bf16(false, a, false, b,
                                                   (short)0, c, false, false);
}

__device__ __forceinline__ float sigmoidf_(float x) {
    return 1.0f / (1.0f + __expf(-x));
}

// LSTM persistent grid: 64 blocks = 2 layers x 32 column-chunks, 4 waves each
#define LSTM_NB 64
#define LSTM_NT 128
// dynamic LDS: Wih slice (4x16x512 bf16) + Whh slice + gate buffer (4x32x16 f32)
#define LSTM_SLICE (4 * 16 * HID)                       // bf16 elements per matrix
#define LSTM_SMEM  (2 * LSTM_SLICE * 2 + 4 * 32 * 16 * 4)

// monotonic grid barrier over LSTM_NB resident blocks (counter zeroed each launch)
__device__ __forceinline__ void grid_barrier(unsigned* __restrict__ bar, int* epoch) {
    ++(*epoch);
    __syncthreads();
    if (threadIdx.x == 0) {
        __threadfence();
        atomicAdd(bar, 1u);
        const unsigned target = (unsigned)(*epoch) * (unsigned)LSTM_NB;
        while (__hip_atomic_load(bar, __ATOMIC_ACQUIRE, __HIP_MEMORY_SCOPE_AGENT) < target) {
            __builtin_amdgcn_s_sleep(1);
        }
    }
    __syncthreads();
}

// ---------------------------------------------------------------------------
// small prep kernels
// ---------------------------------------------------------------------------
__global__ void init_bar_kernel(unsigned* bar) { bar[0] = 0u; }

__global__ void cvt_f32_bf16_kernel(const float* __restrict__ s,
                                    bf16_t* __restrict__ d, int n) {
    int i = blockIdx.x * blockDim.x + threadIdx.x;
    if (i < n) d[i] = (bf16_t)s[i];
}

// x_seq[t][b][:] = emb[token]; token = (t==0) ? inputs[b] : target[b][t-1]
__global__ void embed_gather_kernel(const int* __restrict__ inputs,
                                    const int* __restrict__ tgt,
                                    const float* __restrict__ emb,
                                    bf16_t* __restrict__ xseq) {
    const int tb = blockIdx.x;            // 0 .. TLEN*BATCH-1
    const int t = tb >> 5;                // BATCH == 32
    const int b = tb & 31;
    const int token = (t == 0) ? inputs[b] : tgt[b * TLEN + (t - 1)];
    const float* src = emb + (long)token * HID;
    bf16_t* dst = xseq + (long)tb * HID;
    for (int k = threadIdx.x; k < HID; k += blockDim.x) dst[k] = (bf16_t)src[k];
}

// ---------------------------------------------------------------------------
// persistent pipelined LSTM:
//   block = (layer, 16-col chunk); wave = one gate (i/f/g/o) for that chunk.
//   Weights LDS-resident; layer2 runs one step behind layer1 (65 stages).
// ---------------------------------------------------------------------------
__global__ void __launch_bounds__(LSTM_NT, 1)
lstm_persistent_kernel(const bf16_t* __restrict__ xseq,
                       const bf16_t* __restrict__ Wih1, const bf16_t* __restrict__ Whh1,
                       const bf16_t* __restrict__ Wih2, const bf16_t* __restrict__ Whh2,
                       const float* __restrict__ bih1, const float* __restrict__ bhh1,
                       const float* __restrict__ bih2, const float* __restrict__ bhh2,
                       const float* __restrict__ hiddens,
                       bf16_t* __restrict__ h1buf,   // 2 x (32x512)
                       bf16_t* __restrict__ h2buf,   // 2 x (32x512)
                       float* __restrict__ c1, float* __restrict__ c2,
                       float* __restrict__ h1f,      // 32x512 f32 (final h1)
                       bf16_t* __restrict__ h2all,   // 64 x 32 x 512
                       unsigned* __restrict__ bar) {
    extern __shared__ char smem[];
    bf16_t* wih_s = (bf16_t*)smem;                   // [4*16][512]
    bf16_t* whh_s = wih_s + LSTM_SLICE;              // [4*16][512]
    float*  gate_s = (float*)(whh_s + LSTM_SLICE);   // [4][32][16]

    const int tid = threadIdx.x;
    const int layer = blockIdx.x >> 5;   // 0 or 1
    const int chunk = blockIdx.x & 31;   // 16-column chunk

    const bf16_t* Wih = layer ? Wih2 : Wih1;
    const bf16_t* Whh = layer ? Whh2 : Whh1;
    const float*  bih = layer ? bih2 : bih1;
    const float*  bhh = layer ? bhh2 : bhh1;
    float* C = layer ? c2 : c1;

    // ---- state init (t=0 reads buffer 0 of each h state) ----
    const int gtid = blockIdx.x * LSTM_NT + tid;
    const int gthreads = LSTM_NB * LSTM_NT;
    for (int i = gtid; i < BATCH * HID; i += gthreads) {
        const float hv = hiddens[i];
        h1buf[i] = (bf16_t)hv;
        h1f[i] = hv;
        h2buf[i] = (bf16_t)0.0f;
        c1[i] = 0.0f;
        c2[i] = 0.0f;
    }

    // ---- stage weight slice into LDS (once; reused for all 64 steps) ----
    // slice element i = ((g*16)+r)*512 + k  <->  global row g*512 + chunk*16 + r
    for (int i = tid; i < LSTM_SLICE; i += LSTM_NT) {
        const int g = i >> 13;
        const int r = (i >> 9) & 15;
        const int k = i & 511;
        const long grow = (long)(g * HID + chunk * 16 + r) * HID + k;
        wih_s[i] = Wih[grow];
        whh_s[i] = Whh[grow];
    }

    int epoch = 0;
    grid_barrier(bar, &epoch);   // init + weight staging visible

    const int wave = tid >> 5;           // == gate id (0..3)
    const int lane = tid & 31;
    const int laneM = lane & 15;
    const int mhi = (lane < 16) ? 0 : 8;
    const int col = chunk * 16 + laneM;
    const float bias = bih[wave * HID + col] + bhh[wave * HID + col];

    // ---- pipelined stages: layer1 computes step s, layer2 computes step s-1
#pragma unroll 1
    for (int s = 0; s <= TLEN; ++s) {
        const int t = (layer == 0) ? s : s - 1;
        if (t >= 0 && t < TLEN) {
            // h state produced at step t lives in buffer (t+1)&1
            const bf16_t* xsrc = (layer == 0)
                ? xseq + (long)t * BATCH * HID
                : h1buf + ((t + 1) & 1) * (BATCH * HID);
            bf16_t* hstate = (layer == 0) ? h1buf : h2buf;
            const bf16_t* hsrc = hstate + (t & 1) * (BATCH * HID);
            bf16_t* hdst = hstate + ((t + 1) & 1) * (BATCH * HID);

            v8f acc0 = zero8(), acc1 = zero8();
#pragma unroll
            for (int phase = 0; phase < 2; ++phase) {
                const bf16_t* A = phase ? hsrc : xsrc;
                const bf16_t* Ws = phase ? whh_s : wih_s;
#pragma unroll 2
                for (int k0 = 0; k0 < HID; k0 += 32) {
                    v16bf a0 = load_frag_a(A, laneM, HID, k0, lane);
                    v16bf a1 = load_frag_a(A, 16 + laneM, HID, k0, lane);
                    v16bf b = load_frag_b(Ws, wave * 16 + laneM, HID, k0, lane);
                    acc0 = wmma_bf16(a0, b, acc0);
                    acc1 = wmma_bf16(a1, b, acc1);
                }
            }

            // publish this wave's gate pre-activations (+bias) to LDS
#pragma unroll
            for (int m = 0; m < 2; ++m)
#pragma unroll
                for (int r = 0; r < 8; ++r) {
                    const int row = m * 16 + r + mhi;
                    gate_s[(wave * 32 + row) * 16 + laneM] =
                        (m ? acc1[r] : acc0[r]) + bias;
                }
            __syncthreads();

            // pointwise LSTM update: 32 rows x 16 cols over 128 threads
            for (int i = tid; i < 32 * 16; i += LSTM_NT) {
                const int row = i >> 4;
                const int cc = i & 15;
                const float iv = sigmoidf_(gate_s[(0 * 32 + row) * 16 + cc]);
                const float fv = sigmoidf_(gate_s[(1 * 32 + row) * 16 + cc]);
                const float gv = tanhf(gate_s[(2 * 32 + row) * 16 + cc]);
                const float ov = sigmoidf_(gate_s[(3 * 32 + row) * 16 + cc]);
                const int idx = row * HID + chunk * 16 + cc;
                const float cn = fv * C[idx] + iv * gv;
                C[idx] = cn;
                const float hn = ov * tanhf(cn);
                hdst[idx] = (bf16_t)hn;
                if (layer == 0) h1f[idx] = hn;
                else h2all[(long)t * BATCH * HID + idx] = (bf16_t)hn;
            }
        }
        grid_barrier(bar, &epoch);
    }
}

// ---------------------------------------------------------------------------
// output projection: logits = h2_all (2048x512) @ Wout^T (32000x512) + b_out
// block = 8 waves -> 64(M) x 128(N) tile; wave = 32x32 (4 WMMA accumulators)
// ---------------------------------------------------------------------------
__global__ void __launch_bounds__(256, 1)
proj_kernel(const bf16_t* __restrict__ A,     // 2048 x 512
            const bf16_t* __restrict__ W,     // 32000 x 512
            const float* __restrict__ bout,   // 32000
            float* __restrict__ out) {        // 2048 x 32000
    const int tid = threadIdx.x;
    const int wave = tid >> 5;
    const int lane = tid & 31;
    const int laneM = lane & 15;
    const int mhi = (lane < 16) ? 0 : 8;

    const int mbase = blockIdx.y * 64 + (wave >> 2) * 32;
    const int nbase = blockIdx.x * 128 + (wave & 3) * 32;

    v8f acc[2][2];
#pragma unroll
    for (int mi = 0; mi < 2; ++mi)
#pragma unroll
        for (int ni = 0; ni < 2; ++ni) acc[mi][ni] = zero8();

#pragma unroll 2
    for (int k0 = 0; k0 < HID; k0 += 32) {
        // prefetch next K-tile of the streaming weight rows (global_prefetch_b8)
        if (k0 + 32 < HID) {
            __builtin_prefetch(W + (long)(nbase + laneM) * HID + k0 + 32, 0, 3);
            __builtin_prefetch(W + (long)(nbase + 16 + laneM) * HID + k0 + 32, 0, 3);
        }
        v16bf a0 = load_frag_a(A, mbase + laneM, HID, k0, lane);
        v16bf a1 = load_frag_a(A, mbase + 16 + laneM, HID, k0, lane);
        v16bf b0 = load_frag_b(W, nbase + laneM, HID, k0, lane);
        v16bf b1 = load_frag_b(W, nbase + 16 + laneM, HID, k0, lane);
        acc[0][0] = wmma_bf16(a0, b0, acc[0][0]);
        acc[0][1] = wmma_bf16(a0, b1, acc[0][1]);
        acc[1][0] = wmma_bf16(a1, b0, acc[1][0]);
        acc[1][1] = wmma_bf16(a1, b1, acc[1][1]);
    }

#pragma unroll
    for (int ni = 0; ni < 2; ++ni) {
        const int colv = nbase + ni * 16 + laneM;
        const float bb = bout[colv];
#pragma unroll
        for (int mi = 0; mi < 2; ++mi) {
#pragma unroll
            for (int r = 0; r < 8; ++r) {
                const int row = mbase + mi * 16 + r + mhi;
                out[(long)row * VOCAB + colv] = acc[mi][ni][r] + bb;
            }
        }
    }
}

// in-place row softmax over VOCAB
__global__ void __launch_bounds__(256)
softmax_kernel(float* __restrict__ out) {
    __shared__ float red[256];
    float* p = out + (long)blockIdx.x * VOCAB;
    const int tid = threadIdx.x;

    float mx = -3.402823466e38f;
    for (int i = tid; i < VOCAB; i += 256) mx = fmaxf(mx, p[i]);
    red[tid] = mx;
    __syncthreads();
    for (int s = 128; s > 0; s >>= 1) {
        if (tid < s) red[tid] = fmaxf(red[tid], red[tid + s]);
        __syncthreads();
    }
    mx = red[0];
    __syncthreads();

    float sum = 0.0f;
    for (int i = tid; i < VOCAB; i += 256) sum += __expf(p[i] - mx);
    red[tid] = sum;
    __syncthreads();
    for (int s = 128; s > 0; s >>= 1) {
        if (tid < s) red[tid] += red[tid + s];
        __syncthreads();
    }
    const float inv = 1.0f / red[0];
    __syncthreads();

    for (int i = tid; i < VOCAB; i += 256) p[i] = __expf(p[i] - mx) * inv;
}

__global__ void copy_h1_kernel(const float* __restrict__ h1f, float* __restrict__ dst) {
    const int i = blockIdx.x * blockDim.x + threadIdx.x;
    if (i < BATCH * HID) dst[i] = h1f[i];
}

// ---------------------------------------------------------------------------
// host launcher
// ---------------------------------------------------------------------------
extern "C" void kernel_launch(void* const* d_in, const int* in_sizes, int n_in,
                              void* d_out, int out_size, void* d_ws, size_t ws_size,
                              hipStream_t stream) {
    (void)in_sizes; (void)n_in; (void)out_size; (void)ws_size;
    const int*   inputs  = (const int*)d_in[0];
    const float* hiddens = (const float*)d_in[1];
    const int*   tgt     = (const int*)d_in[2];
    /* d_in[3] = use_tf: reference always teacher-forces; ignored */
    const float* emb  = (const float*)d_in[4];
    const float* Wih1 = (const float*)d_in[5];
    const float* Whh1 = (const float*)d_in[6];
    const float* bih1 = (const float*)d_in[7];
    const float* bhh1 = (const float*)d_in[8];
    const float* Wih2 = (const float*)d_in[9];
    const float* Whh2 = (const float*)d_in[10];
    const float* bih2 = (const float*)d_in[11];
    const float* bhh2 = (const float*)d_in[12];
    const float* Wout = (const float*)d_in[13];
    const float* bout = (const float*)d_in[14];

    char* ws = (char*)d_ws;
    size_t off = 0;
    auto take = [&](size_t bytes) -> char* {
        char* p = ws + off;
        off += (bytes + 255) & ~(size_t)255;
        return p;
    };

    unsigned* bar  = (unsigned*)take(256);
    bf16_t* Wih1b  = (bf16_t*)take((size_t)G4H * HID * 2);
    bf16_t* Whh1b  = (bf16_t*)take((size_t)G4H * HID * 2);
    bf16_t* Wih2b  = (bf16_t*)take((size_t)G4H * HID * 2);
    bf16_t* Whh2b  = (bf16_t*)take((size_t)G4H * HID * 2);
    bf16_t* Woutb  = (bf16_t*)take((size_t)VOCAB * HID * 2);
    bf16_t* xseq   = (bf16_t*)take((size_t)TLEN * BATCH * HID * 2);
    bf16_t* h1buf  = (bf16_t*)take((size_t)2 * BATCH * HID * 2);
    bf16_t* h2buf  = (bf16_t*)take((size_t)2 * BATCH * HID * 2);
    float*  c1     = (float*)take((size_t)BATCH * HID * 4);
    float*  c2     = (float*)take((size_t)BATCH * HID * 4);
    float*  h1f    = (float*)take((size_t)BATCH * HID * 4);
    bf16_t* h2all  = (bf16_t*)take((size_t)TLEN * BATCH * HID * 2);

    float* out = (float*)d_out;

    init_bar_kernel<<<1, 1, 0, stream>>>(bar);

    const int nW = G4H * HID;  // 1,048,576
    cvt_f32_bf16_kernel<<<(nW + 255) / 256, 256, 0, stream>>>(Wih1, Wih1b, nW);
    cvt_f32_bf16_kernel<<<(nW + 255) / 256, 256, 0, stream>>>(Whh1, Whh1b, nW);
    cvt_f32_bf16_kernel<<<(nW + 255) / 256, 256, 0, stream>>>(Wih2, Wih2b, nW);
    cvt_f32_bf16_kernel<<<(nW + 255) / 256, 256, 0, stream>>>(Whh2, Whh2b, nW);
    const int nWo = VOCAB * HID;  // 16,384,000
    cvt_f32_bf16_kernel<<<(nWo + 255) / 256, 256, 0, stream>>>(Wout, Woutb, nWo);

    embed_gather_kernel<<<TLEN * BATCH, 256, 0, stream>>>(inputs, tgt, emb, xseq);

    lstm_persistent_kernel<<<LSTM_NB, LSTM_NT, LSTM_SMEM, stream>>>(
        xseq, Wih1b, Whh1b, Wih2b, Whh2b,
        bih1, bhh1, bih2, bhh2, hiddens,
        h1buf, h2buf, c1, c2, h1f, h2all, bar);

    proj_kernel<<<dim3(VOCAB / 128, (TLEN * BATCH) / 64), 256, 0, stream>>>(
        h2all, Woutb, bout, out);

    softmax_kernel<<<TLEN * BATCH, 256, 0, stream>>>(out);

    copy_h1_kernel<<<(BATCH * HID + 255) / 256, 256, 0, stream>>>(
        h1f, out + (long)TLEN * BATCH * VOCAB);
}
